// CoordinateDecoder_85650237817590
// MI455X (gfx1250) — matrix operand order repaired and checked by reading, not compiled, not run on hardware
//
#include <hip/hip_runtime.h>
#include <hip/hip_bf16.h>
#include <math.h>

typedef __attribute__((ext_vector_type(16))) __bf16 v16bf;
typedef __attribute__((ext_vector_type(8)))  __bf16 v8bf;
typedef __attribute__((ext_vector_type(8)))  float  v8f;
typedef __attribute__((ext_vector_type(4)))  int    v4i;

#define B_     8
#define H_     128
#define W_     128
#define D_     256
#define N_     8192
#define NFREQ  10
#define MLPW   256
#define INDIM  813          // 2 + 4*10 + 3*256 + 3
#define K0     832          // INDIM padded to multiple of 32
#define TILE_M 64
#define HS     272          // hidden-activation LDS row stride (256 + pad)

#if __has_builtin(__builtin_amdgcn_global_load_async_to_lds_b128)
#define HAVE_ASYNC_LDS 1
#endif

// ---- bf16 <-> f32 helpers (RNE) ----
__device__ __forceinline__ __bf16 f2bf(float x) {
    union { float f; unsigned int u; } v; v.f = x;
    unsigned int r = (v.u + 0x7FFFu + ((v.u >> 16) & 1u)) >> 16;
    union { unsigned short s; __bf16 b; } o; o.s = (unsigned short)r;
    return o.b;
}
__device__ __forceinline__ float bf2f(__bf16 b) {
    union { unsigned short s; __bf16 b; } i; i.b = b;
    union { float f; unsigned int u; } v; v.u = ((unsigned int)i.s) << 16;
    return v.f;
}
__device__ __forceinline__ float gelu_f(float x) {
    float x3 = x * x * x;
    float t = tanhf(0.7978845608028654f * (x + 0.044715f * x3));
    return 0.5f * x * (1.0f + t);
}

// 64-byte weight-slab copy per thread: global bf16 -> LDS (async path if available)
__device__ __forceinline__ void copy64B(const __bf16* __restrict__ g, __bf16* __restrict__ l) {
#ifdef HAVE_ASYNC_LDS
    typedef __attribute__((address_space(1))) v4i gv4i_t;   // global int4
    typedef __attribute__((address_space(3))) v4i lv4i_t;   // LDS int4
    gv4i_t* gp = (gv4i_t*)g;
    lv4i_t* lp = (lv4i_t*)l;
    __builtin_amdgcn_global_load_async_to_lds_b128(gp, lp, 0, 0);
    __builtin_amdgcn_global_load_async_to_lds_b128(gp, lp, 16, 0);
    __builtin_amdgcn_global_load_async_to_lds_b128(gp, lp, 32, 0);
    __builtin_amdgcn_global_load_async_to_lds_b128(gp, lp, 48, 0);
#else
    const v8bf* s = (const v8bf*)g;
    v8bf* d = (v8bf*)l;
    d[0] = s[0]; d[1] = s[1]; d[2] = s[2]; d[3] = s[3];
#endif
}
__device__ __forceinline__ void async_fence() {
#ifdef HAVE_ASYNC_LDS
    asm volatile("s_wait_asynccnt 0" ::: "memory");
#endif
}

// ---------------- FiLM params: ctx = cv @ ctx_w + ctx_b ----------------
__global__ void ctx_film_kernel(const float* __restrict__ cv,
                                const float* __restrict__ ctx_w,
                                const float* __restrict__ ctx_b,
                                float* __restrict__ gGamma,
                                float* __restrict__ gBeta) {
    int t = blockIdx.x * blockDim.x + threadIdx.x;
    if (t >= B_ * 2 * MLPW) return;
    int b = t >> 9;
    int j = t & 511;
    float s = ctx_b[j];
    for (int c = 0; c < D_; ++c) s += cv[b * D_ + c] * ctx_w[c * 512 + j];
    if (j < MLPW) gGamma[b * MLPW + j] = s + 1.0f;
    else          gBeta[b * MLPW + (j - MLPW)] = s;
}

// ---- one-time weight conversion f32 -> bf16, col-major [col][k] (B-frag friendly) ----
__global__ void convert_weights_kernel(const float* __restrict__ w0,
                                       const float* __restrict__ wh,
                                       __bf16* __restrict__ wT0,
                                       __bf16* __restrict__ wTh) {
    int idx = blockIdx.x * blockDim.x + threadIdx.x;
    const int tot0 = MLPW * K0;
    if (idx < tot0) {
        int col = idx / K0, k = idx % K0;
        wT0[idx] = f2bf(k < INDIM ? w0[k * MLPW + col] : 0.f);
    }
    int idx2 = idx - tot0;
    if (idx2 >= 0 && idx2 < 3 * MLPW * MLPW) {
        int l = idx2 / (MLPW * MLPW);
        int r = idx2 % (MLPW * MLPW);
        int col = r / MLPW, k = r % MLPW;
        wTh[idx2] = f2bf(wh[((long)l * MLPW + k) * MLPW + col]);
    }
}

// ------- 2x antialiased bilinear downsample (separable [1,3,3,1]/8, edge clamp) -------
__global__ void downsample2x_kernel(const float* __restrict__ src, float* __restrict__ dst,
                                    int hs, int ws) {
    int idx = blockIdx.x * blockDim.x + threadIdx.x;
    int hd = hs >> 1, wd = ws >> 1;
    int total = B_ * hd * wd * D_;
    if (idx >= total) return;
    int c = idx & (D_ - 1);
    int t = idx >> 8;
    int x = t % wd; t /= wd;
    int y = t % hd; t /= hd;
    int b = t;
    const float w4[4] = {0.125f, 0.375f, 0.375f, 0.125f};
    float acc = 0.f;
    #pragma unroll
    for (int dy = 0; dy < 4; ++dy) {
        int sy = 2 * y - 1 + dy; sy = sy < 0 ? 0 : (sy > hs - 1 ? hs - 1 : sy);
        float ra = 0.f;
        #pragma unroll
        for (int dx = 0; dx < 4; ++dx) {
            int sx = 2 * x - 1 + dx; sx = sx < 0 ? 0 : (sx > ws - 1 ? ws - 1 : sx);
            ra += w4[dx] * src[((b * hs + sy) * ws + sx) * D_ + c];
        }
        acc += w4[dy] * ra;
    }
    dst[idx] = acc;
}

// ------- per-level bilinear gather of 64 points into the LDS input tile -------
__device__ __forceinline__ void gather_level(const float* __restrict__ g, int hl,
                                             int b, int pbase, int tid,
                                             const float* __restrict__ coords,
                                             __bf16* __restrict__ sA, int colbase) {
    const int wl = hl;
    for (int p = 0; p < TILE_M; ++p) {
        float cy = coords[2 * (pbase + p) + 0];
        float cx = coords[2 * (pbase + p) + 1];
        float yy = (cy + 1.f) * 0.5f * (float)(hl - 1);
        float xx = (cx + 1.f) * 0.5f * (float)(wl - 1);
        float y0f = fminf(fmaxf(floorf(yy), 0.f), (float)(hl - 1));
        float x0f = fminf(fmaxf(floorf(xx), 0.f), (float)(wl - 1));
        int y0 = (int)y0f, x0 = (int)x0f;
        int y1 = y0 + 1 < hl ? y0 + 1 : hl - 1;
        int x1 = x0 + 1 < wl ? x0 + 1 : wl - 1;
        float wy = yy - y0f, wx = xx - x0f;
        const float* gb = g + ((long)b * hl * wl) * D_ + tid;   // channel = tid, coalesced
        float g00 = gb[(y0 * wl + x0) * D_];
        float g01 = gb[(y0 * wl + x1) * D_];
        float g10 = gb[(y1 * wl + x0) * D_];
        float g11 = gb[(y1 * wl + x1) * D_];
        float top = g00 * (1.f - wx) + g01 * wx;
        float bot = g10 * (1.f - wx) + g11 * wx;
        sA[p * K0 + colbase + tid] = f2bf(top * (1.f - wy) + bot * wy);
    }
}

// ---------------- fused gather + 5-layer FiLM MLP, bf16 WMMA ----------------
__global__ __launch_bounds__(256) void fused_decoder_kernel(
    const float* __restrict__ grid0, const float* __restrict__ grid1,
    const float* __restrict__ grid2,
    const float* __restrict__ coords, const float* __restrict__ oracle,
    const __bf16* __restrict__ wT0, const float* __restrict__ b0,
    const __bf16* __restrict__ wTh, const float* __restrict__ bh,
    const float* __restrict__ ow, const float* __restrict__ ob,
    const float* __restrict__ gGamma, const float* __restrict__ gBeta,
    float* __restrict__ out) {

    extern __shared__ char smem[];
    float*  sGamma = (float*)smem;               // 256 f32
    float*  sBeta  = sGamma + MLPW;              // 256 f32
    __bf16* sW0    = (__bf16*)(sBeta + MLPW);    // 256 cols x 32 k slab, double-buffered
    __bf16* sW1    = sW0 + MLPW * 32;
    __bf16* sA     = sW1 + MLPW * 32;            // 64 x 832 input tile (reused for H ping-pong)
    __bf16* sH0    = sA;
    __bf16* sH1    = sA + TILE_M * HS;

    const int tid  = threadIdx.x;
    const int lane = tid & 31;
    const int wave = tid >> 5;
    const int bid  = blockIdx.x;
    const int b    = bid / (N_ / TILE_M);
    const int ptile = bid % (N_ / TILE_M);
    const int pbase = ptile * TILE_M;

    sGamma[tid] = gGamma[b * MLPW + tid];
    sBeta[tid]  = gBeta[b * MLPW + tid];

    // ---- phase 1a: positional encoding + oracle + zero pad (one point per thread) ----
    if (tid < TILE_M) {
        int p = pbase + tid;
        float cy = coords[2 * p + 0], cx = coords[2 * p + 1];
        __bf16* row = sA + tid * K0;
        row[0] = f2bf(cy); row[1] = f2bf(cx);
        float fr = 3.14159265358979323846f;
        for (int f = 0; f < NFREQ; ++f) {
            float ay = cy * fr, ax = cx * fr;
            row[2 + 4 * f + 0] = f2bf(__sinf(ay));
            row[2 + 4 * f + 1] = f2bf(__sinf(ax));
            row[2 + 4 * f + 2] = f2bf(__cosf(ay));
            row[2 + 4 * f + 3] = f2bf(__cosf(ax));
            fr *= 2.f;
        }
        for (int j = 0; j < 3; ++j)
            row[2 + 4 * NFREQ + 3 * D_ + j] = f2bf(oracle[((long)b * N_ + p) * 3 + j]);
        for (int j = INDIM; j < K0; ++j) row[j] = f2bf(0.f);
    }

    // ---- phase 1b: 3-level bilinear feature gather (channel per thread) ----
    gather_level(grid0, H_,     b, pbase, tid, coords, sA, 2 + 4 * NFREQ + 0 * D_);
    gather_level(grid1, H_ / 2, b, pbase, tid, coords, sA, 2 + 4 * NFREQ + 1 * D_);
    gather_level(grid2, H_ / 4, b, pbase, tid, coords, sA, 2 + 4 * NFREQ + 2 * D_);

    // wave tiling: 4 row-tiles x 2 col-halves; each wave -> 16x128 C tile (8 accumulators)
    const int rt   = wave >> 1;
    const int cb   = (wave & 1) * 128;
    const int rowA = rt * 16 + (lane & 15);
    const int offA = (lane >> 4) * 16;      // A frag: lanes 0-15 K 0..15, 16-31 K 16..31
    const int nloc = lane & 15;
    const int kb0  = (lane >> 4) * 16;      // B frag: lanes 0-15 K 0..15, 16-31 K 16..31

    v8f acc[8];
    #pragma unroll
    for (int j = 0; j < 8; ++j)
        #pragma unroll
        for (int e = 0; e < 8; ++e) acc[j][e] = 0.f;

    // ---- layer 0: (64 x 832) @ (832 x 256), double-buffered weight slabs ----
    copy64B(wT0 + tid * K0, sW0 + tid * 32);            // prologue: slab 0
    for (int kk = 0; kk < K0 / 32; ++kk) {
        const int k = kk * 32;
        async_fence();
        __syncthreads();                                // staged slab kk visible to all
        __bf16* cur = (kk & 1) ? sW1 : sW0;
        if (kk + 1 < K0 / 32)                           // stage slab kk+1 while computing kk
            copy64B(wT0 + tid * K0 + k + 32, ((kk & 1) ? sW0 : sW1) + tid * 32);
        v16bf a;
        {
            const v8bf* pa = (const v8bf*)(sA + rowA * K0 + k + offA);
            v8bf lo = pa[0], hi = pa[1];
            #pragma unroll
            for (int i = 0; i < 8; ++i) { a[i] = lo[i]; a[8 + i] = hi[i]; }
        }
        #pragma unroll
        for (int j = 0; j < 8; ++j) {
            const v8bf* pb = (const v8bf*)(cur + (cb + j * 16 + nloc) * 32 + kb0);
            v8bf blo = pb[0], bhi = pb[1];
            v16bf bb;
            #pragma unroll
            for (int i = 0; i < 8; ++i) { bb[i] = blo[i]; bb[8 + i] = bhi[i]; }
            acc[j] = __builtin_amdgcn_wmma_f32_16x16x32_bf16(
                false, a, false, bb, (short)0, acc[j], false, false);
        }
    }
    __syncthreads();   // sH0 aliases sA: all waves must finish reading sA first

    // ---- layer 0 epilogue: FiLM + GELU -> bf16 activations (reuse sA region) ----
    #pragma unroll
    for (int j = 0; j < 8; ++j) {
        int col = cb + j * 16 + nloc;
        float bias = b0[col];
        float gm = sGamma[col], bt = sBeta[col];
        #pragma unroll
        for (int i = 0; i < 8; ++i) {
            int rowl = rt * 16 + i + ((lane >> 4) << 3);
            float y = (acc[j][i] + bias) * gm + bt;
            sH0[rowl * HS + col] = f2bf(gelu_f(y));
        }
    }
    __syncthreads();

    // ---- hidden layers: 3 x (64 x 256) @ (256 x 256) ----
    __bf16* src = sH0;
    __bf16* dst = sH1;
    for (int l = 0; l < 3; ++l) {
        #pragma unroll
        for (int j = 0; j < 8; ++j)
            #pragma unroll
            for (int e = 0; e < 8; ++e) acc[j][e] = 0.f;

        const __bf16* wl = wTh + (long)l * MLPW * MLPW;  // col-major [col][k]
        copy64B(wl + tid * MLPW, sW0 + tid * 32);        // prologue: slab 0
        for (int kk = 0; kk < MLPW / 32; ++kk) {
            const int k = kk * 32;
            async_fence();
            __syncthreads();
            __bf16* cur = (kk & 1) ? sW1 : sW0;
            if (kk + 1 < MLPW / 32)
                copy64B(wl + tid * MLPW + k + 32, ((kk & 1) ? sW0 : sW1) + tid * 32);
            v16bf a;
            {
                const v8bf* pa = (const v8bf*)(src + rowA * HS + k + offA);
                v8bf lo = pa[0], hi = pa[1];
                #pragma unroll
                for (int i = 0; i < 8; ++i) { a[i] = lo[i]; a[8 + i] = hi[i]; }
            }
            #pragma unroll
            for (int j = 0; j < 8; ++j) {
                const v8bf* pb = (const v8bf*)(cur + (cb + j * 16 + nloc) * 32 + kb0);
                v8bf blo = pb[0], bhi = pb[1];
                v16bf bb;
                #pragma unroll
                for (int i = 0; i < 8; ++i) { bb[i] = blo[i]; bb[8 + i] = bhi[i]; }
                acc[j] = __builtin_amdgcn_wmma_f32_16x16x32_bf16(
                    false, a, false, bb, (short)0, acc[j], false, false);
            }
        }
        #pragma unroll
        for (int j = 0; j < 8; ++j) {
            int col = cb + j * 16 + nloc;
            float bias = bh[l * MLPW + col];
            float gm = sGamma[col], bt = sBeta[col];
            #pragma unroll
            for (int i = 0; i < 8; ++i) {
                int rowl = rt * 16 + i + ((lane >> 4) << 3);
                float y = (acc[j][i] + bias) * gm + bt;
                dst[rowl * HS + col] = f2bf(gelu_f(y));
            }
        }
        __syncthreads();
        __bf16* tmp = src; src = dst; dst = tmp;
    }

    // ---- output head: tanh(H @ out_w + out_b), 64 rows x 3 ----
    if (tid < TILE_M * 3) {
        int rowl = tid / 3, o = tid % 3;
        float s = ob[o];
        const __bf16* hrow = src + rowl * HS;
        for (int c = 0; c < MLPW; ++c) s += bf2f(hrow[c]) * ow[c * 3 + o];
        out[((long)b * N_ + pbase + rowl) * 3 + o] = tanhf(s);
    }
}

extern "C" void kernel_launch(void* const* d_in, const int* in_sizes, int n_in,
                              void* d_out, int out_size, void* d_ws, size_t ws_size,
                              hipStream_t stream) {
    const float* feature_grid   = (const float*)d_in[0];
    const float* context_vector = (const float*)d_in[1];
    const float* coords         = (const float*)d_in[2];
    const float* oracle         = (const float*)d_in[3];
    const float* mlp0_w         = (const float*)d_in[4];
    const float* mlp0_b         = (const float*)d_in[5];
    const float* mlp_hw         = (const float*)d_in[6];
    const float* mlp_hb         = (const float*)d_in[7];
    const float* ctx_w          = (const float*)d_in[8];
    const float* ctx_b          = (const float*)d_in[9];
    const float* out_w          = (const float*)d_in[10];
    const float* out_b          = (const float*)d_in[11];
    float* out = (float*)d_out;

    float* ws = (float*)d_ws;
    float* gGamma = ws;                                   // 8*256
    float* gBeta  = gGamma + B_ * MLPW;                   // 8*256
    float* level1 = gBeta + B_ * MLPW;                    // 8*64*64*256  (33.5 MB)
    float* level2 = level1 + (long)B_ * 64 * 64 * D_;     // 8*32*32*256  ( 8.4 MB)
    __bf16* wT0   = (__bf16*)(level2 + (long)B_ * 32 * 32 * D_);  // 256x832 bf16
    __bf16* wTh   = wT0 + MLPW * K0;                              // 3x256x256 bf16

    ctx_film_kernel<<<(B_ * 2 * MLPW + 255) / 256, 256, 0, stream>>>(
        context_vector, ctx_w, ctx_b, gGamma, gBeta);

    int totw = MLPW * K0 + 3 * MLPW * MLPW;
    convert_weights_kernel<<<(totw + 255) / 256, 256, 0, stream>>>(mlp0_w, mlp_hw, wT0, wTh);

    int tot1 = B_ * 64 * 64 * D_;
    downsample2x_kernel<<<(tot1 + 255) / 256, 256, 0, stream>>>(feature_grid, level1, H_, W_);
    int tot2 = B_ * 32 * 32 * D_;
    downsample2x_kernel<<<(tot2 + 255) / 256, 256, 0, stream>>>(level1, level2, 64, 64);

    size_t smem = (size_t)(2 * MLPW) * 4          // gamma/beta
                + (size_t)(2 * MLPW * 32) * 2     // double-buffered weight slab
                + (size_t)(TILE_M * K0) * 2;      // input tile / activation ping-pong
    fused_decoder_kernel<<<B_ * (N_ / TILE_M), 256, smem, stream>>>(
        feature_grid, level1, level2, coords, oracle,
        wT0, mlp0_b, wTh, mlp_hb, out_w, out_b, gGamma, gBeta, out);
}
